// CustomMamba_89885075570941
// MI455X (gfx1250) — matrix operand order repaired and checked by reading, not compiled, not run on hardware
//
#include <hip/hip_runtime.h>
#include <stdint.h>

// ---------------------------------------------------------------------------
// Types
// ---------------------------------------------------------------------------
typedef __attribute__((ext_vector_type(16))) __bf16   v16bf;
typedef __attribute__((ext_vector_type(8)))  float    v8f;
typedef __attribute__((ext_vector_type(4))) unsigned  u32x4;
typedef __attribute__((ext_vector_type(8)))  int      i32x8;
typedef __attribute__((ext_vector_type(4)))  int      i32x4;

#if defined(__has_builtin)
#  if __has_builtin(__builtin_amdgcn_tensor_load_to_lds)
#    define HAVE_TDM 1
#  endif
#endif
#ifndef HAVE_TDM
#  define HAVE_TDM 0
#endif

#define MROWS 16384   // B*L = 8*2048
#define LSEQ  2048
#define DM    768
#define DI    1536
#define DS    16
#define DTR   48
#define DTPAD 64

// float -> bf16 round-to-nearest-even
static __device__ __forceinline__ unsigned short f2bf(float f) {
  unsigned u = __float_as_uint(f);
  u += 0x7fffu + ((u >> 16) & 1u);
  return (unsigned short)(u >> 16);
}
static __device__ __forceinline__ float siluf(float x) {
  return x / (1.0f + __expf(-x));
}

#if HAVE_TDM
// Issue one 2D TDM tile load: tile_rows x 64 bf16 elements, row stride
// `stride_elems`, destination LDS byte offset `lds_off` (workgroup base).
static __device__ __forceinline__ void tdm_load_tile(
    const unsigned short* gsrc, unsigned lds_off, int stride_elems,
    int tile_rows) {
  unsigned long long ga = (unsigned long long)(uintptr_t)gsrc;
  u32x4 g0;
  g0[0] = 1u;                                    // count = 1 (valid D#)
  g0[1] = lds_off;                               // lds_addr (bytes)
  g0[2] = (unsigned)(ga & 0xffffffffu);          // global_addr[31:0]
  g0[3] = (unsigned)((ga >> 32) & 0x01ffffffu)   // global_addr[56:32]
        | (2u << 30);                            // type = 2 ("image")
  i32x8 g1;
  g1[0] = (1 << 16);                             // data_size = 1 (2 bytes)
  g1[1] = (int)(((unsigned)stride_elems & 0xffffu) << 16);   // dim0 lo16
  g1[2] = (int)(((unsigned)stride_elems >> 16)               // dim0 hi16
        | (((unsigned)tile_rows & 0xffffu) << 16));          // dim1 lo16
  g1[3] = (int)(64u << 16);                      // dim1 hi16=0 | tile_dim0=64
  g1[4] = tile_rows;                             // tile_dim1 | tile_dim2=0
  g1[5] = stride_elems;                          // tensor_dim0_stride lo32
  g1[6] = 0;
  g1[7] = 0;
  i32x4 gz = {0, 0, 0, 0};
#if __clang_major__ >= 23
  i32x8 gz8 = {0, 0, 0, 0, 0, 0, 0, 0};
  __builtin_amdgcn_tensor_load_to_lds(g0, g1, gz, gz, gz8, 0);
#else
  __builtin_amdgcn_tensor_load_to_lds(g0, g1, gz, gz, 0);
#endif
}
#endif

// ---------------------------------------------------------------------------
// GEMM: C[M,N](f32) = A[M,K](bf16, row major) * W[Npad,K](bf16, row major)^T
//
// Workgroup tile: 128 rows x 64 cols, 8 waves, each wave = 16 rows with four
// 16x16 f32 accumulators. K stepped by 64 (8 v_wmma_f32_16x16x32_bf16 between
// barriers). A (128x64) and W (64x64) tiles are fetched by the Tensor Data
// Mover into double-buffered LDS; next tiles' DMAs are issued before waiting
// (s_wait_tensorcnt 2) so HBM latency overlaps WMMA. W is pre-padded to a
// multiple of 64 rows, so the inner loop has no guards / per-lane global
// loads. The TDM writes LDS behind the compiler's back, so a memory-clobber
// asm makes those writes visible to alias analysis (otherwise the fragment
// ds_loads get folded to undef — observed in codegen).
// Requires: K % 64 == 0, M % 128 == 0, grid.y covers Npad/64.
// ---------------------------------------------------------------------------
#define SA_ELEMS (128 * 64)
#define SW_ELEMS (64 * 64)

__global__ void __launch_bounds__(256)
gemm_bf16_wmma(const unsigned short* __restrict__ A,
               const unsigned short* __restrict__ W,
               float* __restrict__ C,
               int N, int K, int ldc)
{
  // Single shared block at LDS offset 0 so TDM descriptor offsets are exact.
  __shared__ __align__(16) unsigned short smem[2 * SA_ELEMS + 2 * SW_ELEMS];

  const int tid  = threadIdx.x;
  const int wv   = tid >> 5;
  const int lane = tid & 31;
  const int m0   = blockIdx.x * 128;
  const int n0   = blockIdx.y * 64;

  v8f acc[4] = {};

  const int rowA = wv * 16 + (lane & 15);  // A-frag row within tile
  const int hi   = lane >> 4;              // lane group 0/1
  const int nl   = lane & 15;              // B-frag column
  const int koff = hi ? 8 : 0;             // B-frag K sub-offset

#if HAVE_TDM
  if (tid < 32) {  // one wave drives the DMA engine (EXEC ignored by TDM)
    tdm_load_tile(A + (size_t)m0 * K, 0u, K, 128);
    tdm_load_tile(W + (size_t)n0 * K, (unsigned)(2 * SA_ELEMS * 2), K, 64);
  }
#endif

  int buf = 0;
  for (int kb = 0; kb < K; kb += 64, buf ^= 1) {
    const unsigned aoff = buf ? (unsigned)(SA_ELEMS * 2) : 0u;
    const unsigned woff = buf ? (unsigned)((2 * SA_ELEMS + SW_ELEMS) * 2)
                              : (unsigned)(2 * SA_ELEMS * 2);
#if HAVE_TDM
    if (tid < 32) {
      if (kb + 64 < K) {
        // Prefetch next tiles into the other buffer, then wait only for the
        // current buffer's two loads (TDM completes in order per wave).
        const unsigned aoff2 = buf ? 0u : (unsigned)(SA_ELEMS * 2);
        const unsigned woff2 = buf ? (unsigned)(2 * SA_ELEMS * 2)
                                   : (unsigned)((2 * SA_ELEMS + SW_ELEMS) * 2);
        tdm_load_tile(A + (size_t)m0 * K + kb + 64, aoff2, K, 128);
        tdm_load_tile(W + (size_t)n0 * K + kb + 64, woff2, K, 64);
        __builtin_amdgcn_s_wait_tensorcnt(2);
      } else {
        __builtin_amdgcn_s_wait_tensorcnt(0);
      }
    }
#else
    {  // cooperative fallback load of current buffer
      unsigned short* dA = smem + (aoff >> 1);
      unsigned short* dW = smem + (woff >> 1);
      for (int c = tid; c < 1024; c += 256) {  // 128 rows x 8 x 16B
        int r = c >> 3, s = c & 7;
        *(u32x4*)(dA + r * 64 + s * 8) =
            *(const u32x4*)(A + (size_t)(m0 + r) * K + kb + s * 8);
      }
      for (int c = tid; c < 512; c += 256) {   // 64 rows x 8 x 16B
        int r = c >> 3, s = c & 7;
        *(u32x4*)(dW + r * 64 + s * 8) =
            *(const u32x4*)(W + (size_t)(n0 + r) * K + kb + s * 8);
      }
    }
#endif
    // The TDM engine wrote smem; tell the compiler LDS changed here so the
    // fragment loads below are not folded away as reads of undef memory.
    asm volatile("" : : "r"(&smem[0]) : "memory");
    __syncthreads();

    const unsigned short* bA = smem + (aoff >> 1);
    const unsigned short* bW = smem + (woff >> 1);

#pragma unroll
    for (int h = 0; h < 2; h++) {  // two K=32 halves of the 64-wide tile
      union { u32x4 q[2]; v16bf v; } af;
      af.q[0] = *(const u32x4*)(bA + rowA * 64 + h * 32 + hi * 8);
      af.q[1] = *(const u32x4*)(bA + rowA * 64 + h * 32 + 16 + hi * 8);
#pragma unroll
      for (int nt = 0; nt < 4; nt++) {
        union { u32x4 q[2]; v16bf v; } bfm;
        const unsigned short* wrow = bW + (nt * 16 + nl) * 64 + h * 32 + koff;
        bfm.q[0] = *(const u32x4*)(wrow);
        bfm.q[1] = *(const u32x4*)(wrow + 16);
        acc[nt] = __builtin_amdgcn_wmma_f32_16x16x32_bf16(
            false, af.v, false, bfm.v, (short)0, acc[nt], false, false);
      }
    }
    __syncthreads();
  }

  // Store: lane holds (N = nl, M = r + 8*hi) for VGPR r of each accumulator.
#pragma unroll
  for (int nt = 0; nt < 4; nt++) {
    int col = n0 + nt * 16 + nl;
    if (col < N) {
#pragma unroll
      for (int r = 0; r < 8; r++) {
        int mrow = m0 + wv * 16 + r + 8 * hi;
        __builtin_nontemporal_store(acc[nt][r], &C[(size_t)mrow * ldc + col]);
      }
    }
  }
}

// ---------------------------------------------------------------------------
// Elementwise / reduction kernels
// ---------------------------------------------------------------------------
__global__ void embed_time_kernel(const int* __restrict__ ids,
                                  const float* __restrict__ times,
                                  const float* __restrict__ embed,
                                  const float* __restrict__ tw,
                                  const float* __restrict__ tb,
                                  float* __restrict__ h) {
  int i = blockIdx.x * 256 + threadIdx.x;  // over M*768 (exact multiple)
  int m = i / DM, d = i - m * DM;
  h[i] = embed[(size_t)ids[m] * DM + d] + times[m] * tw[d] + tb[d];
}

// f32 [rows][cols] -> bf16 [prows][pcols], zero padded.
__global__ void convert_pad_kernel(const float* __restrict__ src,
                                   unsigned short* __restrict__ dst,
                                   int rows, int cols, int prows, int pcols) {
  int i = blockIdx.x * 256 + threadIdx.x;
  if (i >= prows * pcols) return;
  int r = i / pcols, c = i - r * pcols;
  dst[i] = (r < rows && c < cols) ? f2bf(src[(size_t)r * cols + c])
                                  : (unsigned short)0;
}

__global__ void rmsnorm_bf16_kernel(const float* __restrict__ x,
                                    const float* __restrict__ w,
                                    unsigned short* __restrict__ out) {
  int row = blockIdx.x;
  const float* xr = x + (size_t)row * DM;
  float s = 0.f;
  for (int i = threadIdx.x; i < DM; i += 256) { float v = xr[i]; s += v * v; }
  __shared__ float red[256];
  red[threadIdx.x] = s;
  __syncthreads();
  for (int off = 128; off > 0; off >>= 1) {
    if (threadIdx.x < off) red[threadIdx.x] += red[threadIdx.x + off];
    __syncthreads();
  }
  float scale = rsqrtf(red[0] / (float)DM + 1e-5f);
  for (int i = threadIdx.x; i < DM; i += 256)
    out[(size_t)row * DM + i] = f2bf(xr[i] * scale * w[i]);
}

__global__ void conv_silu_kernel(const float* __restrict__ xz,
                                 const float* __restrict__ cw,
                                 const float* __restrict__ cb,
                                 float* __restrict__ xc,
                                 unsigned short* __restrict__ xcb) {
  int i = blockIdx.x * 256 + threadIdx.x;  // over M*1536
  int d = i % DI, m = i / DI;
  int l = m & (LSEQ - 1);
  float acc = cb[d];
#pragma unroll
  for (int j = 0; j < 4; j++) {
    int ls = l - 3 + j;
    if (ls >= 0) acc += cw[d * 4 + j] * xz[(size_t)(m - 3 + j) * (2 * DI) + d];
  }
  float s = siluf(acc);
  xc[i] = s;
  xcb[i] = f2bf(s);
}

__global__ void split_dt_kernel(const float* __restrict__ xdbl,  // [M][80]
                                unsigned short* __restrict__ dtb16) {  // [M][64]
  int i = blockIdx.x * 256 + threadIdx.x;  // M*64
  int m = i >> 6, c = i & 63;
  dtb16[i] = (c < DTR) ? f2bf(xdbl[(size_t)m * 80 + c]) : (unsigned short)0;
}

__global__ void softplus_bias_kernel(float* __restrict__ delta,
                                     const float* __restrict__ dtb) {
  int i = blockIdx.x * 256 + threadIdx.x;  // M*1536
  int d = i % DI;
  float x = delta[i] + dtb[d];
  delta[i] = (x > 20.f) ? x : log1pf(__expf(x));
}

// Sequential selective scan: one thread per (b, d), 16 states in registers.
__global__ void scan_kernel(const float* __restrict__ xc,     // u  [M][1536]
                            const float* __restrict__ delta,  // [M][1536]
                            const float* __restrict__ xdbl,   // [M][80]
                            const float* __restrict__ A_log,  // [1536][16]
                            const float* __restrict__ Dv,     // [1536]
                            float* __restrict__ y) {          // [M][1536]
  int b = blockIdx.x / (DI / 256);
  int dblk = blockIdx.x % (DI / 256);
  int d = dblk * 256 + threadIdx.x;
  float Ad[DS];
#pragma unroll
  for (int n = 0; n < DS; n++) Ad[n] = -__expf(A_log[d * DS + n]);
  float Dd = Dv[d];
  float h[DS];
#pragma unroll
  for (int n = 0; n < DS; n++) h[n] = 0.f;
  __shared__ float sBC[32];
  for (int t = 0; t < LSEQ; t++) {
    size_t m = (size_t)b * LSEQ + t;
    if (threadIdx.x < 32) sBC[threadIdx.x] = xdbl[m * 80 + DTR + threadIdx.x];
    __syncthreads();
    float dl = delta[m * DI + d];
    float u  = xc[m * DI + d];
    float du = dl * u;
    float acc = 0.f;
#pragma unroll
    for (int n = 0; n < DS; n++) {
      h[n] = h[n] * __expf(dl * Ad[n]) + du * sBC[n];
      acc += h[n] * sBC[16 + n];
    }
    y[m * DI + d] = acc + u * Dd;
    __syncthreads();
  }
}

__global__ void gate_kernel(const float* __restrict__ y,
                            const float* __restrict__ xz,
                            unsigned short* __restrict__ ygb) {
  int i = blockIdx.x * 256 + threadIdx.x;  // M*1536
  int d = i % DI, m = i / DI;
  float z = xz[(size_t)m * (2 * DI) + DI + d];
  ygb[i] = f2bf(y[i] * siluf(z));
}

__global__ void residual_kernel(float* __restrict__ h,
                                const float* __restrict__ yo) {
  int i = blockIdx.x * 256 + threadIdx.x;  // M*768
  h[i] += yo[i];
}

// ---------------------------------------------------------------------------
// Host: orchestration
// ---------------------------------------------------------------------------
extern "C" void kernel_launch(void* const* d_in, const int* in_sizes, int n_in,
                              void* d_out, int out_size, void* d_ws, size_t ws_size,
                              hipStream_t stream) {
  const int M = MROWS;
  const int*   ids     = (const int*)d_in[0];
  const float* times   = (const float*)d_in[1];
  const float* embed   = (const float*)d_in[3];
  const float* time_w  = (const float*)d_in[4];
  const float* time_b  = (const float*)d_in[5];
  const float* in_w    = (const float*)d_in[6];   // [4][3072][768]
  const float* conv_w  = (const float*)d_in[7];   // [4][1536][4]
  const float* conv_b  = (const float*)d_in[8];   // [4][1536]
  const float* xp_w    = (const float*)d_in[9];   // [4][80][1536]
  const float* dt_w    = (const float*)d_in[10];  // [4][1536][48]
  const float* dt_b    = (const float*)d_in[11];  // [4][1536]
  const float* A_log   = (const float*)d_in[12];  // [4][1536][16]
  const float* D_skip  = (const float*)d_in[13];  // [4][1536]
  const float* out_w   = (const float*)d_in[14];  // [4][768][1536]
  const float* norm_w  = (const float*)d_in[15];  // [4][768]
  const float* normf_w = (const float*)d_in[16];  // [768]
  float* logits = (float*)d_out;                  // [M][1969]

  char* ws = (char*)d_ws;
  size_t off = 0;
  auto carve = [&](size_t bytes) -> void* {
    void* p = ws + off;
    off += (bytes + 255) & ~(size_t)255;
    return p;
  };
  float*          hbuf  = (float*)carve((size_t)M * DM * 4);
  unsigned short* nrm   = (unsigned short*)carve((size_t)M * DM * 2);
  float*          xz    = (float*)carve((size_t)M * 2 * DI * 4);
  float*          xc    = (float*)carve((size_t)M * DI * 4);
  unsigned short* xcb   = (unsigned short*)carve((size_t)M * DI * 2);
  float*          xdbl  = (float*)carve((size_t)M * 80 * 4);
  unsigned short* dtb16 = (unsigned short*)carve((size_t)M * DTPAD * 2);
  float*          delta = (float*)carve((size_t)M * DI * 4);
  float*          ybuf  = (float*)carve((size_t)M * DI * 4);
  unsigned short* ygb   = (unsigned short*)carve((size_t)M * DI * 2);
  float*          yout  = (float*)carve((size_t)M * DM * 4);
  unsigned short* w_in  = (unsigned short*)carve((size_t)2 * DI * DM * 2);
  unsigned short* w_x   = (unsigned short*)carve((size_t)128 * DI * 2);  // 80->128
  unsigned short* w_dt  = (unsigned short*)carve((size_t)DI * DTPAD * 2);
  unsigned short* w_out = (unsigned short*)carve((size_t)DM * DI * 2);
  unsigned short* emb16 = (unsigned short*)carve((size_t)1984 * DM * 2); // 1969->1984

  auto gemm = [&](const unsigned short* A, const unsigned short* W, float* C,
                  int N, int Npad, int K, int ldc) {
    dim3 g(M / 128, Npad / 64);
    gemm_bf16_wmma<<<g, 256, 0, stream>>>(A, W, C, N, K, ldc);
  };
  auto convp = [&](const float* s, unsigned short* d, int rows, int cols,
                   int prows, int pcols) {
    int n = prows * pcols;
    convert_pad_kernel<<<(n + 255) / 256, 256, 0, stream>>>(s, d, rows, cols,
                                                            prows, pcols);
  };

  // h = embed[ids] + times*time_w + time_b
  embed_time_kernel<<<(M * DM) / 256, 256, 0, stream>>>(ids, times, embed,
                                                        time_w, time_b, hbuf);
  convp(embed, emb16, 1969, DM, 1984, DM);

  for (int i = 0; i < 4; i++) {
    rmsnorm_bf16_kernel<<<M, 256, 0, stream>>>(hbuf, norm_w + i * DM, nrm);
    convp(in_w + (size_t)i * 2 * DI * DM, w_in, 2 * DI, DM, 2 * DI, DM);
    convp(xp_w + (size_t)i * 80 * DI, w_x, 80, DI, 128, DI);
    convp(dt_w + (size_t)i * DI * DTR, w_dt, DI, DTR, DI, DTPAD);
    convp(out_w + (size_t)i * DM * DI, w_out, DM, DI, DM, DI);

    // xz = normed @ in_proj^T   [M,3072]
    gemm(nrm, w_in, xz, 2 * DI, 2 * DI, DM, 2 * DI);
    // x = silu(causal_conv1d(xz[:, :1536]))
    conv_silu_kernel<<<(M * DI) / 256, 256, 0, stream>>>(
        xz, conv_w + (size_t)i * DI * 4, conv_b + (size_t)i * DI, xc, xcb);
    // x_dbl = x @ x_proj^T   [M,80]
    gemm(xcb, w_x, xdbl, 80, 128, DI, 80);
    split_dt_kernel<<<(M * DTPAD) / 256, 256, 0, stream>>>(xdbl, dtb16);
    // delta_raw = dt @ dt_proj^T   [M,1536]
    gemm(dtb16, w_dt, delta, DI, DI, DTPAD, DI);
    softplus_bias_kernel<<<(M * DI) / 256, 256, 0, stream>>>(
        delta, dt_b + (size_t)i * DI);
    // selective scan
    scan_kernel<<<8 * (DI / 256), 256, 0, stream>>>(
        xc, delta, xdbl, A_log + (size_t)i * DI * DS, D_skip + (size_t)i * DI,
        ybuf);
    gate_kernel<<<(M * DI) / 256, 256, 0, stream>>>(ybuf, xz, ygb);
    // out = (y*silu(z)) @ out_proj^T   [M,768]
    gemm(ygb, w_out, yout, DM, DM, DI, DM);
    residual_kernel<<<(M * DM) / 256, 256, 0, stream>>>(hbuf, yout);
  }

  rmsnorm_bf16_kernel<<<M, 256, 0, stream>>>(hbuf, normf_w, nrm);
  // logits = normed @ embed^T   [M,1969]
  gemm(nrm, emb16, logits, 1969, 1984, DM, 1969);
}
